// GauntTensorProduct_74526272520554
// MI455X (gfx1250) — compile-verified, matrix operand order checked by reading, not run on hardware
//
#include <hip/hip_runtime.h>

#define NLM 25
#define NP 240          // 16 beta * 15 alpha grid points
#define FEAT_W 3200     // 128 * 25

typedef __attribute__((ext_vector_type(16))) __bf16 v16bf;
typedef __attribute__((ext_vector_type(8)))  float  v8f;
typedef __attribute__((ext_vector_type(4)))  int    v4i;

union FragB16 { v16bf v; uint4 u[2]; unsigned short s[16]; };
union Acc8    { v8f v; float f[8]; };

__constant__ int c_LIDX[25] = {0,1,1,1,2,2,2,2,2,3,3,3,3,3,3,3,4,4,4,4,4,4,4,4,4};

// native bf16 conversion (RNE) -> lets ISel use hardware cvt instead of integer emulation
static __device__ __forceinline__ unsigned short f2bf(float x) {
    union { __bf16 h; unsigned short s; } c;
    c.h = (__bf16)x;
    return c.s;
}

// ---- CDNA5 async global->LDS staging (ASYNCcnt path), with portable fallback ----
#if defined(__AMDGCN__) && __has_builtin(__builtin_amdgcn_global_load_async_to_lds_b128)
#define HAVE_ASYNC_LDS 1
typedef __attribute__((address_space(1))) v4i GV4;
typedef __attribute__((address_space(3))) v4i LV4;
#else
#define HAVE_ASYNC_LDS 0
#endif

static __device__ __forceinline__ void stage_b128(const uint4* __restrict__ g,
                                                  uint4* __restrict__ l,
                                                  int count, int tid, int nthreads) {
#if HAVE_ASYNC_LDS
    for (int j = tid; j < count; j += nthreads)
        __builtin_amdgcn_global_load_async_to_lds_b128((GV4*)(g + j), (LV4*)(l + j), 0, 0);
#else
    for (int j = tid; j < count; j += nthreads)
        l[j] = g[j];
#endif
}

static __device__ __forceinline__ void stage_wait() {
#if HAVE_ASYNC_LDS
    asm volatile("s_wait_asynccnt 0" ::: "memory");
#endif
}

static __device__ void gauss16(int g, double* xo, double* wo) {
    const double PI = 3.14159265358979323846;
    int k = 16 - g;                       // roots descending; g ascending
    double x = cos(PI * (k - 0.25) / 16.5);
    for (int it = 0; it < 60; ++it) {
        double p0 = 1.0, p1 = x;
        for (int n = 1; n < 16; ++n) { double p2 = ((2*n+1)*x*p1 - n*p0)/(n+1); p0 = p1; p1 = p2; }
        double dp = 16.0*(x*p1 - p0)/(x*x - 1.0);
        double dx = p1/dp;
        x -= dx;
        if (fabs(dx) < 1e-15) break;
    }
    double p0 = 1.0, p1 = x;
    for (int n = 1; n < 16; ++n) { double p2 = ((2*n+1)*x*p1 - n*p0)/(n+1); p0 = p1; p1 = p2; }
    double dp = 16.0*(x*p1 - p0)/(x*x - 1.0);
    *xo = x;
    *wo = 2.0/((1.0 - x*x)*dp*dp);
}

// ---------------- setup A: Y (f32, 25x240) and quadrature weights ----------------
__global__ void setupA(float* __restrict__ g_Yf, float* __restrict__ g_wq) {
    const double PI = 3.14159265358979323846;
    int p = threadIdx.x;
    if (p < 16) {
        double x, wgt; gauss16(p, &x, &wgt);
        g_wq[p] = (float)(wgt * (2.0*PI/15.0));
    }
    if (p < NP) {
        int g = p / 15, h = p % 15;
        double x, wgt; gauss16(g, &x, &wgt);
        double alpha = 2.0*PI*h/15.0;
        double P[5][5];
        for (int a = 0; a < 5; ++a) for (int b = 0; b < 5; ++b) P[a][b] = 0.0;
        P[0][0] = 1.0;
        double s = sqrt(fmax(1.0 - x*x, 0.0));
        for (int m = 1; m <= 4; ++m) P[m][m] = -(2*m-1)*s*P[m-1][m-1];
        for (int m = 0; m < 4; ++m)  P[m+1][m] = (2*m+1)*x*P[m][m];
        for (int m = 0; m <= 4; ++m)
            for (int l = m+2; l <= 4; ++l)
                P[l][m] = ((2*l-1)*x*P[l-1][m] - (l+m-1)*P[l-2][m]) / (l-m);
        double fact[9] = {1,1,2,6,24,120,720,5040,40320};
        int i = 0;
        for (int l = 0; l <= 4; ++l) {
            for (int m = -l; m <= l; ++m) {
                int am = m < 0 ? -m : m;
                double K = sqrt((2*l+1)/(4.0*PI) * fact[l-am]/fact[l+am]);
                double v;
                if (m == 0)      v = K*P[l][0];
                else if (m > 0)  v = sqrt(2.0)*K*P[l][am]*cos(am*alpha);
                else             v = sqrt(2.0)*K*P[l][am]*sin(am*alpha);
                g_Yf[i*NP + p] = (float)v;
                ++i;
            }
        }
    }
}

// ------- setup B: bf16 WMMA-B-fragment swizzles of Y (25->32 x 240) and Yw^T (240->256 x 25->32) -------
__global__ void setupB(const float* __restrict__ g_Yf, const float* __restrict__ g_wq,
                       unsigned short* __restrict__ g_YBfrag, unsigned short* __restrict__ g_YwTfrag) {
    int tid = threadIdx.x;
    // [nt(15)][lane(32)][e(16)] : value Y[k][N], k = kmap(e,khalf), N = nt*16+(lane&15)
    for (int idx = tid; idx < 15*32*16; idx += 256) {
        int e = idx & 15, lane = (idx >> 4) & 31, nt = idx >> 9;
        int khalf = lane >> 4, N = nt*16 + (lane & 15);
        int k = ((e & 8) << 1) + (khalf << 3) + (e & 7);
        float v = (k < NLM) ? g_Yf[k*NP + N] : 0.0f;
        g_YBfrag[idx] = f2bf(v);
    }
    // [ks(8)][nt2(2)][lane(32)][e(16)] : value Y[i][p]*wq[p/15], i = nt2*16+(lane&15), p = ks*32+kmap
    for (int idx = tid; idx < 8*2*32*16; idx += 256) {
        int e = idx & 15, lane = (idx >> 4) & 31, nt2 = (idx >> 9) & 1, ks = idx >> 10;
        int khalf = lane >> 4, i = nt2*16 + (lane & 15);
        int p = ks*32 + ((e & 8) << 1) + (khalf << 3) + (e & 7);
        float v = (i < NLM && p < NP) ? g_Yf[i*NP + p] * g_wq[p/15] : 0.0f;
        g_YwTfrag[idx] = f2bf(v);
    }
}

// ------- setup C: wc (5,128,128) f32 -> bf16 B-fragment swizzle [l][ks(4)][nt(8)][lane(32)][e(16)] -------
__global__ void setupC(const float* __restrict__ wc, unsigned short* __restrict__ g_wcfrag) {
    int idx = blockIdx.x*256 + threadIdx.x;
    if (idx >= 5*4*8*32*16) return;
    int e = idx & 15, lane = (idx >> 4) & 31, nt = (idx >> 9) & 7, ks = (idx >> 12) & 3, l = idx >> 14;
    int khalf = lane >> 4;
    int a = nt*16 + (lane & 15);
    int c = ks*32 + ((e & 8) << 1) + (khalf << 3) + (e & 7);
    g_wcfrag[idx] = f2bf(wc[(l*128 + c)*128 + a]);
}

// ------- kernel 1: per-b fused GEMM1 (image@Y), x2-scale, GEMM2 (@Yw^T) -> Feat[b][i][c] bf16 -------
__global__ void k_fused(const float* __restrict__ feature,
                        const float* __restrict__ sh,
                        const float* __restrict__ w1,
                        const float* __restrict__ w2,
                        const float* __restrict__ g_Yf,
                        const unsigned short* __restrict__ g_YBfrag,
                        const unsigned short* __restrict__ g_YwTfrag,
                        unsigned short* __restrict__ g_Feat) {
    extern __shared__ char smem[];
    unsigned short* sYB  = (unsigned short*)smem;         // 7680
    unsigned short* sYwT = sYB  + 7680;                   // 8192
    unsigned short* sImg = sYwT + 8192;                   // 128*32
    unsigned short* sX3  = sImg + 128*32;                 // 128*264 (bf16, padded K=256 + bank pad)
    float* sX2 = (float*)(sX3 + 128*264);                 // 240
    float* sKer = sX2 + 240;                              // 25

    const int tid  = threadIdx.x;
    const int b    = blockIdx.x;
    const int lane = tid & 31, w = tid >> 5;
    const int khalf = lane >> 4, l15 = lane & 15;

    // stage Y fragments (L2-hot constants) via async global->LDS
    stage_b128((const uint4*)g_YBfrag,  (uint4*)sYB,  7680/8, tid, 256);
    stage_b128((const uint4*)g_YwTfrag, (uint4*)sYwT, 8192/8, tid, 256);

    // image[c][i] = feature[b, l-block] * w1[c][l], bf16, K padded to 32
    const float* frow = feature + (long)b * FEAT_W;
    for (int j = tid; j < 128*32; j += 256) {
        int c = j >> 5, ii = j & 31;
        float v = 0.0f;
        if (ii < NLM) {
            int l = c_LIDX[ii];
            v = frow[128*l*l + c*(2*l+1) + (ii - l*l)] * w1[c*5 + l];
        }
        sImg[j] = f2bf(v);
    }
    // zero the K-pad columns 240..263 of sX3
    for (int j = tid; j < 128*24; j += 256) sX3[(j/24)*264 + 240 + (j%24)] = 0;
    if (tid < NLM) sKer[tid] = sh[b*NLM + tid] * w2[c_LIDX[tid]];
    stage_wait();
    __syncthreads();

    // x2[p] = sum_i kernel_i * Y[i][p]
    if (tid < NP) {
        float acc = 0.0f;
#pragma unroll
        for (int i = 0; i < NLM; ++i) acc += sKer[i] * g_Yf[i*NP + tid];
        sX2[tid] = acc;
    }
    __syncthreads();

    // GEMM1: X1 = image @ Y  (M=128, N=240, K=32), scale by x2, store bf16
    {
        FragB16 afrag;
        const unsigned short* arow = sImg + (w*16 + l15)*32;
        afrag.u[0] = *(const uint4*)(arow + 8*khalf);
        afrag.u[1] = *(const uint4*)(arow + 16 + 8*khalf);
        for (int nt = 0; nt < 15; ++nt) {
            FragB16 bfrag;
            const unsigned short* bsrc = sYB + (nt*32 + lane)*16;
            bfrag.u[0] = *(const uint4*)bsrc;
            bfrag.u[1] = *(const uint4*)(bsrc + 8);
            Acc8 acc;
#pragma unroll
            for (int e = 0; e < 8; ++e) acc.f[e] = 0.0f;
            acc.v = __builtin_amdgcn_wmma_f32_16x16x32_bf16(false, afrag.v, false, bfrag.v,
                                                            (short)0, acc.v, false, false);
            int p = nt*16 + l15;
            float xs = sX2[p];
            int crow = w*16 + 8*khalf;
#pragma unroll
            for (int e = 0; e < 8; ++e)
                sX3[(crow + e)*264 + p] = f2bf(acc.f[e] * xs);
        }
    }
    __syncthreads();

    // GEMM2: Feat = X3 @ Yw^T (M=128, N=32, K=256): A-frag loaded once, dual accumulators
    {
        Acc8 acc0, acc1;
#pragma unroll
        for (int e = 0; e < 8; ++e) { acc0.f[e] = 0.0f; acc1.f[e] = 0.0f; }
#pragma unroll
        for (int ks = 0; ks < 8; ++ks) {
            FragB16 afrag, b0, b1;
            const unsigned short* arow = sX3 + (w*16 + l15)*264 + ks*32;
            afrag.u[0] = *(const uint4*)(arow + 8*khalf);
            afrag.u[1] = *(const uint4*)(arow + 16 + 8*khalf);
            const unsigned short* bs0 = sYwT + ((ks*2 + 0)*32 + lane)*16;
            const unsigned short* bs1 = sYwT + ((ks*2 + 1)*32 + lane)*16;
            b0.u[0] = *(const uint4*)bs0;  b0.u[1] = *(const uint4*)(bs0 + 8);
            b1.u[0] = *(const uint4*)bs1;  b1.u[1] = *(const uint4*)(bs1 + 8);
            acc0.v = __builtin_amdgcn_wmma_f32_16x16x32_bf16(false, afrag.v, false, b0.v,
                                                             (short)0, acc0.v, false, false);
            acc1.v = __builtin_amdgcn_wmma_f32_16x16x32_bf16(false, afrag.v, false, b1.v,
                                                             (short)0, acc1.v, false, false);
        }
        const int cbase = w*16 + 8*khalf;
        {   // nt2 = 0 : i = l15 in [0,16) -> always valid
            unsigned short tmp[8];
#pragma unroll
            for (int e = 0; e < 8; ++e) tmp[e] = f2bf(acc0.f[e]);
            *(uint4*)(g_Feat + ((long)b*NLM + l15)*128 + cbase) = *(const uint4*)tmp;
        }
        if (l15 < 9) {  // nt2 = 1 : i = 16 + l15 valid while < 25
            unsigned short tmp[8];
#pragma unroll
            for (int e = 0; e < 8; ++e) tmp[e] = f2bf(acc1.f[e]);
            *(uint4*)(g_Feat + ((long)b*NLM + 16 + l15)*128 + cbase) = *(const uint4*)tmp;
        }
    }
}

// ------- kernel 2: per (b-tile, i) channel mix: Out[b,a,i] = sum_c Feat[b,i,c] * wc[l][c][a] -------
__global__ void k_mix(const unsigned short* __restrict__ g_Feat,
                      const unsigned short* __restrict__ g_wcfrag,
                      float* __restrict__ out) {
    extern __shared__ char smem[];
    unsigned short* sA = (unsigned short*)smem;  // 128x128 bf16, row-major [b_local][c]
    unsigned short* sB = sA + 128*128;           // 16384 bf16, frag-swizzled wc[l]

    const int tid = threadIdx.x;
    const int bt  = blockIdx.x;
    const int i   = blockIdx.y;
    const int l   = c_LIDX[i];
    const int d   = 2*l + 1, r = i - l*l;
    const int lane = tid & 31, w = tid >> 5;
    const int khalf = lane >> 4, l15 = lane & 15;

    // A rows: 16 b128 transfers per row, row-strided in global
    for (int j = tid; j < 128*16; j += 256) {
        int row = j >> 4, q = j & 15;
        stage_b128(((const uint4*)(g_Feat + ((long)(bt*128 + row)*NLM + i)*128)) + q,
                   ((uint4*)sA) + row*16 + q, 1, 0, 1);
    }
    stage_b128((const uint4*)(g_wcfrag + l*16384), (uint4*)sB, 2048, tid, 256);
    stage_wait();
    __syncthreads();

    FragB16 bfr[4];
#pragma unroll
    for (int ks = 0; ks < 4; ++ks) {
        const unsigned short* bsrc = sB + ((ks*8 + w)*32 + lane)*16;
        bfr[ks].u[0] = *(const uint4*)bsrc;
        bfr[ks].u[1] = *(const uint4*)(bsrc + 8);
    }
    const long outb0 = (long)(bt*128) * FEAT_W + 128*l*l + r;
    for (int mt = 0; mt < 8; ++mt) {
        Acc8 acc;
#pragma unroll
        for (int e = 0; e < 8; ++e) acc.f[e] = 0.0f;
#pragma unroll
        for (int ks = 0; ks < 4; ++ks) {
            FragB16 afr;
            const unsigned short* arow = sA + (mt*16 + l15)*128 + ks*32;
            afr.u[0] = *(const uint4*)(arow + 8*khalf);
            afr.u[1] = *(const uint4*)(arow + 16 + 8*khalf);
            acc.v = __builtin_amdgcn_wmma_f32_16x16x32_bf16(false, afr.v, false, bfr[ks].v,
                                                            (short)0, acc.v, false, false);
        }
        int a = w*16 + l15;
        int brow = mt*16 + 8*khalf;
#pragma unroll
        for (int e = 0; e < 8; ++e)
            out[outb0 + (long)(brow + e)*FEAT_W + a*d] = acc.f[e];
    }
}

extern "C" void kernel_launch(void* const* d_in, const int* in_sizes, int n_in,
                              void* d_out, int out_size, void* d_ws, size_t ws_size,
                              hipStream_t stream) {
    (void)in_sizes; (void)n_in; (void)out_size; (void)ws_size;
    const float* feature = (const float*)d_in[0];
    const float* sh      = (const float*)d_in[1];
    const float* w1      = (const float*)d_in[2];
    const float* w2      = (const float*)d_in[3];
    const float* wc      = (const float*)d_in[4];
    float* out = (float*)d_out;

    char* ws = (char*)d_ws;
    size_t off = 0;
    auto carve = [&](size_t bytes) -> char* {
        char* p = ws + off;
        off = (off + bytes + 255) & ~(size_t)255;
        return p;
    };
    float* g_Yf = (float*)carve(25*240*4);
    float* g_wq = (float*)carve(16*4);
    unsigned short* g_YBfrag  = (unsigned short*)carve(7680*2);
    unsigned short* g_YwTfrag = (unsigned short*)carve(8192*2);
    unsigned short* g_wcfrag  = (unsigned short*)carve(81920*2);
    unsigned short* g_Feat    = (unsigned short*)carve((size_t)16384*3200*2);

    setupA<<<1, 256, 0, stream>>>(g_Yf, g_wq);
    setupB<<<1, 256, 0, stream>>>(g_Yf, g_wq, g_YBfrag, g_YwTfrag);
    setupC<<<320, 256, 0, stream>>>(wc, g_wcfrag);

    size_t smem1 = (size_t)(7680 + 8192 + 128*32 + 128*264)*2 + (240 + 32)*4;
    k_fused<<<16384, 256, smem1, stream>>>(feature, sh, w1, w2, g_Yf,
                                           g_YBfrag, g_YwTfrag, g_Feat);

    size_t smem2 = (size_t)(128*128 + 16384)*2;
    k_mix<<<dim3(128, 25), 256, smem2, stream>>>(g_Feat, g_wcfrag, out);
}
